// HeteroGAT_49976239456884
// MI455X (gfx1250) — matrix operand order repaired and checked by reading, not compiled, not run on hardware
//
#include <hip/hip_runtime.h>
#include <hip/hip_bf16.h>
#include <math.h>

// ---------------------------------------------------------------------------
// HeteroGAT on MI455X (gfx1250, wave32).
//   xs = x_src @ W           (WMMA f16->f32 GEMM; f16 operands pre-converted,
//                             B pre-packed to WMMA fragment layout and staged
//                             into LDS via global_load_async_to_lds_b128)
//   a_src = <xs, att_src> ;  a_dst = x_dst @ (W @ att_dst)  (folded)
//   segment softmax over edges (atomic max, atomic exp-sum)
//   scatter: out[dst] += (alpha_h / H) * xs[src][h]  (head mean folded in,
//                             float4 coalesced gathers)
//   finalize: +bias, LayerNorm, ReLU  (one wave per node, shfl reductions)
// ---------------------------------------------------------------------------

typedef __attribute__((ext_vector_type(16))) _Float16 v16h;
typedef __attribute__((ext_vector_type(8)))  _Float16 v8h;
typedef __attribute__((ext_vector_type(8)))  float    v8f;

#define HH 2
#define CC 64
#define DD 64
#define NK 128   // H*C

// ---------------------------------------------------------------- utilities
__global__ void fill_kernel(float* __restrict__ p, float v, int n) {
    int i = blockIdx.x * blockDim.x + threadIdx.x;
    if (i < n) p[i] = v;
}

__global__ void cvt_f16_kernel(const float* __restrict__ x,
                               _Float16* __restrict__ y, int n) {
    int i = blockIdx.x * blockDim.x + threadIdx.x;
    if (i < n) y[i] = (_Float16)x[i];
}

__device__ inline void atomicMaxF(float* addr, float val) {
    if (val >= 0.0f) atomicMax((int*)addr, __float_as_int(val));
    else             atomicMin((unsigned int*)addr, __float_as_uint(val));
}

// ------------------------------------------------- wa_dst[d][h] = sum_c W[d][h*C+c]*att_dst[h][c]
__global__ void make_wadst_kernel(const float* __restrict__ W,
                                  const float* __restrict__ att_dst,
                                  float* __restrict__ wa) {
    int t = threadIdx.x;            // 0..127 ; t = d*2 + h
    int d = t >> 1, h = t & 1;
    float s = 0.0f;
    #pragma unroll 8
    for (int c = 0; c < CC; ++c)
        s += W[d * NK + h * CC + c] * att_dst[h * CC + c];
    wa[t] = s;
}

// ------------------------------------------------- pack W into WMMA B-fragment layout (f16)
// Wp[((kc*8 + nt)*32 + lane)*16 + t] = W[(kc*32 + (t>>3)*16 + (lane>>4)*8 + (t&7))*128 + nt*16 + (lane&15)]
__global__ void pack_w_kernel(const float* __restrict__ W,
                              _Float16* __restrict__ Wp) {
    int i = blockIdx.x * blockDim.x + threadIdx.x;   // 0..8191
    if (i >= 2 * 8 * 32 * 16) return;
    int t    = i & 15;
    int lane = (i >> 4) & 31;
    int nt   = (i >> 9) & 7;
    int kc   = i >> 12;
    int k    = kc * 32 + (t >> 3) * 16 + (lane >> 4) * 8 + (t & 7);
    int col  = nt * 16 + (lane & 15);
    Wp[i] = (_Float16)W[(size_t)k * NK + col];
}

// ------------------------------------------------- WMMA GEMM: Y[M,128] = X[M,64] @ W[64,128]
// One wave per 16-row slab. Packed B fragments (16 KB) async-DMA'd to LDS.
__global__ __launch_bounds__(256)
void gemm_xw_wmma_kernel(const _Float16* __restrict__ Xh,
                         const _Float16* __restrict__ Wp,
                         float* __restrict__ Y, int M) {
    __shared__ __align__(16) _Float16 ldsW[8192];   // 16 KB: [kc][nt][lane][16]

    const int tid  = threadIdx.x;
    const int lane = tid & 31;
    const int wave = tid >> 5;

    // ---- CDNA5 async DMA: stage packed B fragments into LDS (ASYNCcnt) ----
    {
        unsigned base = (unsigned)(uintptr_t)&ldsW[0];
        #pragma unroll
        for (int it = 0; it < 4; ++it) {
            unsigned off = (unsigned)(tid * 16 + it * 4096);
            unsigned long long g = (unsigned long long)(uintptr_t)Wp + off;
            asm volatile("global_load_async_to_lds_b128 %0, %1, off"
                         :: "v"(base + off), "v"(g)
                         : "memory");
        }
        asm volatile("s_wait_asynccnt 0x0" ::: "memory");
    }
    __syncthreads();

    const int mt = blockIdx.x * 8 + wave;   // 16-row tile index
    const int m0 = mt * 16;
    if (m0 >= M) return;                    // wave-uniform; EXEC all-1s for WMMA

    const int l15 = lane & 15;
    const int hi  = lane >> 4;              // 0: lanes 0-15, 1: lanes 16-31

    // A tile (16x64 as two 16x32 chunks), ISA 16-bit A layout; four b128 loads.
    //  lanes 0-15 : element t -> K = (t<8 ? t : t+8)
    //  lanes 16-31: element t -> K = (t<8 ? t+8 : t+16)
    const _Float16* xrow = Xh + (size_t)(m0 + l15) * DD + hi * 8;
    v8h a00 = *(const v8h*)(xrow + 0);
    v8h a01 = *(const v8h*)(xrow + 16);
    v8h a10 = *(const v8h*)(xrow + 32);
    v8h a11 = *(const v8h*)(xrow + 48);
    v16h a0 = __builtin_shufflevector(a00, a01, 0,1,2,3,4,5,6,7,8,9,10,11,12,13,14,15);
    v16h a1 = __builtin_shufflevector(a10, a11, 0,1,2,3,4,5,6,7,8,9,10,11,12,13,14,15);

    #pragma unroll
    for (int nt = 0; nt < 8; ++nt) {
        const int n0 = nt * 16;
        const int f0 = ((0 * 8 + nt) * 32 + lane) * 16;   // k-chunk 0 fragment
        const int f1 = ((1 * 8 + nt) * 32 + lane) * 16;   // k-chunk 1 fragment
        v8h b0lo = *(const v8h*)&ldsW[f0];
        v8h b0hi = *(const v8h*)&ldsW[f0 + 8];
        v8h b1lo = *(const v8h*)&ldsW[f1];
        v8h b1hi = *(const v8h*)&ldsW[f1 + 8];
        v16h b0 = __builtin_shufflevector(b0lo, b0hi, 0,1,2,3,4,5,6,7,8,9,10,11,12,13,14,15);
        v16h b1 = __builtin_shufflevector(b1lo, b1hi, 0,1,2,3,4,5,6,7,8,9,10,11,12,13,14,15);

        v8f c = {};
        c = __builtin_amdgcn_wmma_f32_16x16x32_f16(false, a0, false, b0,
                                                   (short)0, c, false, false);
        c = __builtin_amdgcn_wmma_f32_16x16x32_f16(false, a1, false, b1,
                                                   (short)0, c, false, false);
        // D layout: lanes 0-15 -> M=v, N=lane ; lanes 16-31 -> M=v+8, N=lane-16
        #pragma unroll
        for (int v = 0; v < 8; ++v) {
            int row = m0 + hi * 8 + v;
            Y[(size_t)row * NK + n0 + l15] = c[v];
        }
    }
}

// ------------------------------------------------- a_src[n,h] = sum_c xs[n,h,c]*att_src[h,c]
__global__ void attn_src_kernel(const float* __restrict__ xs,
                                const float* __restrict__ att,
                                float* __restrict__ a, int N) {
    int i = blockIdx.x * blockDim.x + threadIdx.x;   // over N*H
    if (i >= N * HH) return;
    int n = i >> 1, h = i & 1;
    const float* p = xs + (size_t)n * NK + h * CC;
    const float* q = att + h * CC;
    float s = 0.0f;
    #pragma unroll 8
    for (int c = 0; c < CC; ++c) s += p[c] * q[c];
    a[i] = s;
}

// ------------------------------------------------- a_dst[n,h] = sum_d x[n,d]*wa[d,h]
__global__ void attn_dst_kernel(const float* __restrict__ x,
                                const float* __restrict__ wa,
                                float* __restrict__ a, int N) {
    int i = blockIdx.x * blockDim.x + threadIdx.x;
    if (i >= N * HH) return;
    int n = i >> 1, h = i & 1;
    const float* p = x + (size_t)n * DD;
    float s = 0.0f;
    #pragma unroll 8
    for (int d = 0; d < DD; ++d) s += p[d] * wa[d * HH + h];
    a[i] = s;
}

// ------------------------------------------------- segment max over edges
__global__ void edge_max_kernel(const int* __restrict__ ei,
                                const float* __restrict__ asrc,
                                const float* __restrict__ adst,
                                float* __restrict__ m, int E) {
    int e = blockIdx.x * blockDim.x + threadIdx.x;
    if (e >= E) return;
    int s = ei[e], d = ei[E + e];
    #pragma unroll
    for (int h = 0; h < HH; ++h) {
        float v = asrc[s * HH + h] + adst[d * HH + h];
        v = v > 0.0f ? v : 0.2f * v;         // leaky_relu
        atomicMaxF(&m[d * HH + h], v);
    }
}

// ------------------------------------------------- segment exp-sum
__global__ void edge_sum_kernel(const int* __restrict__ ei,
                                const float* __restrict__ asrc,
                                const float* __restrict__ adst,
                                const float* __restrict__ m,
                                float* __restrict__ denom, int E) {
    int e = blockIdx.x * blockDim.x + threadIdx.x;
    if (e >= E) return;
    int s = ei[e], d = ei[E + e];
    #pragma unroll
    for (int h = 0; h < HH; ++h) {
        float v = asrc[s * HH + h] + adst[d * HH + h];
        v = v > 0.0f ? v : 0.2f * v;
        atomicAdd(&denom[d * HH + h], __expf(v - m[d * HH + h]));
    }
}

// ------------------------------------------------- message scatter (head mean folded in)
// One wave per edge; each lane gathers one float4 of xs (coalesced b128).
__global__ __launch_bounds__(256)
void edge_scatter_kernel(const int* __restrict__ ei,
                         const float* __restrict__ asrc,
                         const float* __restrict__ adst,
                         const float* __restrict__ m,
                         const float* __restrict__ denom,
                         const float* __restrict__ xs,
                         float* __restrict__ acc, int E) {
    int lane = threadIdx.x & 31;
    int e = blockIdx.x * 8 + (threadIdx.x >> 5);
    if (e >= E) return;
    int s = ei[e], d = ei[E + e];
    float al[HH];
    #pragma unroll
    for (int h = 0; h < HH; ++h) {
        float v = asrc[s * HH + h] + adst[d * HH + h];
        v = v > 0.0f ? v : 0.2f * v;
        al[h] = __expf(v - m[d * HH + h]) / denom[d * HH + h] * (1.0f / HH);
    }
    const float4* xr4 = (const float4*)(xs + (size_t)s * NK);
    float4 xv = xr4[lane];                 // channels c0..c0+3, one head each
    int c0 = lane * 4;
    float a = al[c0 >> 6];
    float* ob = acc + (size_t)d * CC + (c0 & (CC - 1));
    atomicAdd(ob + 0, a * xv.x);
    atomicAdd(ob + 1, a * xv.y);
    atomicAdd(ob + 2, a * xv.z);
    atomicAdd(ob + 3, a * xv.w);
}

// ------------------------------------------------- +bias, LayerNorm, ReLU (in place)
__global__ __launch_bounds__(256)
void ln_relu_kernel(float* __restrict__ acc,
                    const float* __restrict__ bias,
                    const float* __restrict__ gamma,
                    const float* __restrict__ beta, int N) {
    int lane = threadIdx.x & 31;
    int n = blockIdx.x * 8 + (threadIdx.x >> 5);
    if (n >= N) return;
    float* p = acc + (size_t)n * CC;
    float v0 = p[lane]      + bias[lane];
    float v1 = p[lane + 32] + bias[lane + 32];
    float s  = v0 + v1;
    float s2 = v0 * v0 + v1 * v1;
    #pragma unroll
    for (int off = 16; off > 0; off >>= 1) {
        s  += __shfl_xor(s,  off, 32);
        s2 += __shfl_xor(s2, off, 32);
    }
    float mu  = s * (1.0f / CC);
    float var = s2 * (1.0f / CC) - mu * mu;
    float r   = rsqrtf(var + 1e-5f);
    float y0  = (v0 - mu) * r * gamma[lane]      + beta[lane];
    float y1  = (v1 - mu) * r * gamma[lane + 32] + beta[lane + 32];
    p[lane]      = y0 > 0.0f ? y0 : 0.0f;
    p[lane + 32] = y1 > 0.0f ? y1 : 0.0f;
}

// ---------------------------------------------------------------------------
extern "C" void kernel_launch(void* const* d_in, const int* in_sizes, int n_in,
                              void* d_out, int out_size, void* d_ws, size_t ws_size,
                              hipStream_t stream) {
    const float* x_user   = (const float*)d_in[0];
    const float* x_item   = (const float*)d_in[1];
    const int*   ei_u2i   = (const int*)  d_in[2];
    const int*   ei_i2u   = (const int*)  d_in[3];
    const float* W_u2i    = (const float*)d_in[4];
    const float* as_u2i   = (const float*)d_in[5];
    const float* ad_u2i   = (const float*)d_in[6];
    const float* b_u2i    = (const float*)d_in[7];
    const float* W_i2u    = (const float*)d_in[8];
    const float* as_i2u   = (const float*)d_in[9];
    const float* ad_i2u   = (const float*)d_in[10];
    const float* b_i2u    = (const float*)d_in[11];
    const float* g_user   = (const float*)d_in[12];
    const float* be_user  = (const float*)d_in[13];
    const float* g_item   = (const float*)d_in[14];
    const float* be_item  = (const float*)d_in[15];

    const int Nu = in_sizes[0] / DD;       // 100000
    const int Ni = in_sizes[1] / DD;       // 100000
    const int E1 = in_sizes[2] / 2;        // 1000000 (u2i)
    const int E2 = in_sizes[3] / 2;        // 1000000 (i2u)

    // ------- workspace layout -------
    float* ws = (float*)d_ws;
    float* xs_u2i    = ws; ws += (size_t)Nu * NK;   // x_user @ W_u2i
    float* xs_i2u    = ws; ws += (size_t)Ni * NK;   // x_item @ W_i2u
    float* asrc_u2i  = ws; ws += (size_t)Nu * HH;
    float* adst_u2i  = ws; ws += (size_t)Ni * HH;
    float* asrc_i2u  = ws; ws += (size_t)Ni * HH;
    float* adst_i2u  = ws; ws += (size_t)Nu * HH;
    float* m_u2i     = ws; ws += (size_t)Ni * HH;   // m_* contiguous
    float* m_i2u     = ws; ws += (size_t)Nu * HH;
    float* den_u2i   = ws; ws += (size_t)Ni * HH;   // denom_* contiguous
    float* den_i2u   = ws; ws += (size_t)Nu * HH;
    float* wa_u2i    = ws; ws += DD * HH;
    float* wa_i2u    = ws; ws += DD * HH;
    // 16-byte aligned half-precision region
    uintptr_t hb = ((uintptr_t)ws + 15) & ~(uintptr_t)15;
    _Float16* xh_user = (_Float16*)hb;               // [Nu][64] f16
    _Float16* xh_item = xh_user + (size_t)Nu * DD;   // [Ni][64] f16
    _Float16* wp_u2i  = xh_item + (size_t)Ni * DD;   // 8192 packed fragments
    _Float16* wp_i2u  = wp_u2i + 8192;

    float* acc_user = (float*)d_out;                    // [Nu][C]
    float* acc_item = (float*)d_out + (size_t)Nu * CC;  // [Ni][C]

    // ------- init: out accum = 0, softmax max = -inf, denom = 0 -------
    {
        int n = (Nu + Ni) * CC;
        fill_kernel<<<(n + 255) / 256, 256, 0, stream>>>((float*)d_out, 0.0f, n);
        int nm = (Ni + Nu) * HH;
        fill_kernel<<<(nm + 255) / 256, 256, 0, stream>>>(m_u2i, -INFINITY, nm);
        fill_kernel<<<(nm + 255) / 256, 256, 0, stream>>>(den_u2i, 0.0f, nm);
    }

    // ------- operand conversion / packing -------
    cvt_f16_kernel<<<(Nu * DD + 255) / 256, 256, 0, stream>>>(x_user, xh_user, Nu * DD);
    cvt_f16_kernel<<<(Ni * DD + 255) / 256, 256, 0, stream>>>(x_item, xh_item, Ni * DD);
    pack_w_kernel<<<32, 256, 0, stream>>>(W_u2i, wp_u2i);
    pack_w_kernel<<<32, 256, 0, stream>>>(W_i2u, wp_i2u);
    make_wadst_kernel<<<1, 128, 0, stream>>>(W_u2i, ad_u2i, wa_u2i);
    make_wadst_kernel<<<1, 128, 0, stream>>>(W_i2u, ad_i2u, wa_i2u);

    // ------- WMMA GEMMs: xs = x_src @ W -------
    {
        int tilesU = (Nu + 15) / 16, tilesI = (Ni + 15) / 16;
        gemm_xw_wmma_kernel<<<(tilesU + 7) / 8, 256, 0, stream>>>(xh_user, wp_u2i, xs_u2i, Nu);
        gemm_xw_wmma_kernel<<<(tilesI + 7) / 8, 256, 0, stream>>>(xh_item, wp_i2u, xs_i2u, Ni);
    }

    // ------- attention scores -------
    attn_src_kernel<<<(Nu * HH + 255) / 256, 256, 0, stream>>>(xs_u2i, as_u2i, asrc_u2i, Nu);
    attn_src_kernel<<<(Ni * HH + 255) / 256, 256, 0, stream>>>(xs_i2u, as_i2u, asrc_i2u, Ni);
    attn_dst_kernel<<<(Ni * HH + 255) / 256, 256, 0, stream>>>(x_item, wa_u2i, adst_u2i, Ni);
    attn_dst_kernel<<<(Nu * HH + 255) / 256, 256, 0, stream>>>(x_user, wa_i2u, adst_i2u, Nu);

    // ------- segment softmax: max then exp-sum -------
    edge_max_kernel<<<(E1 + 255) / 256, 256, 0, stream>>>(ei_u2i, asrc_u2i, adst_u2i, m_u2i, E1);
    edge_max_kernel<<<(E2 + 255) / 256, 256, 0, stream>>>(ei_i2u, asrc_i2u, adst_i2u, m_i2u, E2);
    edge_sum_kernel<<<(E1 + 255) / 256, 256, 0, stream>>>(ei_u2i, asrc_u2i, adst_u2i, m_u2i, den_u2i, E1);
    edge_sum_kernel<<<(E2 + 255) / 256, 256, 0, stream>>>(ei_i2u, asrc_i2u, adst_i2u, m_i2u, den_i2u, E2);

    // ------- weighted message scatter (one wave per edge) -------
    edge_scatter_kernel<<<(E1 + 7) / 8, 256, 0, stream>>>(ei_u2i, asrc_u2i, adst_u2i,
                                                          m_u2i, den_u2i, xs_u2i, acc_item, E1);
    edge_scatter_kernel<<<(E2 + 7) / 8, 256, 0, stream>>>(ei_i2u, asrc_i2u, adst_i2u,
                                                          m_i2u, den_i2u, xs_i2u, acc_user, E2);

    // ------- finalize: bias + LayerNorm + ReLU -------
    ln_relu_kernel<<<(Nu + 7) / 8, 256, 0, stream>>>(acc_user, b_i2u, g_user, be_user, Nu);
    ln_relu_kernel<<<(Ni + 7) / 8, 256, 0, stream>>>(acc_item, b_u2i, g_item, be_item, Ni);
}
